// MultiHeadAttention_17824114278699
// MI455X (gfx1250) — matrix operand order repaired and checked by reading, not compiled, not run on hardware
//
#include <hip/hip_runtime.h>

// ---------------------------------------------------------------------------
// MultiHeadAttention (softmax over QUERY axis) for MI455X / gfx1250.
//   K0: fp32 -> bf16 conversions (QUERY_SCALE folded into queries).
//   K1: bf16 WMMA GEMM Y = X @ W^T with async-to-LDS double-buffered W tile.
//   K2: S[b,h,k] = sum_q exp(logits[q,k]); one wave per (b,h,ktile), dp
//       fragments hoisted, no atomics.
//   K4: vsT[b][e][k] = dp[b][k][e] / S (transposed for contiguous B-frags).
//   K5: block-cooperative flash-style context: 8 waves share async-staged,
//       double-buffered dp (128x64) and vsT (64x128) LDS tiles per k-step;
//       P = exp(logits) built with WMMA + exp, relaid D->A through per-wave
//       LDS, then WMMA'd against the vsT tile.  Cuts L2 traffic 8x vs the
//       per-wave version so the kernel is WMMA-bound, not L2-bound.
// Working set (~42 MB) is L2-resident (192 MB); ~47 GFLOP all in bf16 WMMA.
// ---------------------------------------------------------------------------

#define DEV static __device__ __forceinline__

typedef __bf16 bf16;
typedef bf16  v16bf __attribute__((ext_vector_type(16)));
typedef bf16  v8bf  __attribute__((ext_vector_type(8)));
typedef float v8f   __attribute__((ext_vector_type(8)));
typedef int   v4i   __attribute__((ext_vector_type(4)));

constexpr int   B_  = 4;
constexpr int   LQ_ = 1024;
constexpr int   LD_ = 4096;
constexpr int   D_  = 512;
constexpr int   H_  = 8;
constexpr int   DH_ = 64;
constexpr float QUERY_SCALE_ = 0.125f;   // (512/8)^-0.5

DEV int lane_id() { return (int)(threadIdx.x & 31u); }

// A-fragment (16x32 bf16, MxK), row-major source, leading dim `ld`.
DEV v16bf load_frag_a(const bf16* base, int ld) {
  const int l = lane_id();
  const int m = l & 15, hi = l >> 4;
  const bf16* p = base + (size_t)m * ld + hi * 8;
  v8bf c0 = *(const v8bf*)p;
  v8bf c1 = *(const v8bf*)(p + 16);
  v16bf r;
#pragma unroll
  for (int i = 0; i < 8; ++i) { r[i] = c0[i]; r[i + 8] = c1[i]; }
  return r;
}

// B-fragment (32x16 bf16, KxN) where column n of B is a contiguous row of
// the row-major source (B[k][n] = W[n0+n][k0+k]).
DEV v16bf load_frag_b(const bf16* base, int ld) {
  const int l = lane_id();
  const int n = l & 15, kh = l >> 4;
  const bf16* p = base + (size_t)n * ld + kh * 8;
  v8bf c0 = *(const v8bf*)p;
  v8bf c1 = *(const v8bf*)(p + 16);
  v16bf r;
#pragma unroll
  for (int i = 0; i < 8; ++i) { r[i] = c0[i]; r[i + 8] = c1[i]; }
  return r;
}

DEV v8f wmma_bf16(v16bf a, v16bf b, v8f c) {
  return __builtin_amdgcn_wmma_f32_16x16x32_bf16(
      false, a, false, b, (short)0, c, false, false);
}

DEV void store_d_f32(float* base, int ld, v8f d) {
  const int l = lane_id();
  const int n = l & 15, mh = (l >> 4) * 8;
#pragma unroll
  for (int r = 0; r < 8; ++r) base[(size_t)(mh + r) * ld + n] = d[r];
}

DEV void store_d_bf16(bf16* base, int ld, v8f d) {
  const int l = lane_id();
  const int n = l & 15, mh = (l >> 4) * 8;
#pragma unroll
  for (int r = 0; r < 8; ++r) base[(size_t)(mh + r) * ld + n] = (bf16)d[r];
}

// 16B async copy global -> LDS (ASYNCcnt-tracked, per CDNA5 ISA).
DEV void async_copy16(bf16* lds_dst, const bf16* gsrc) {
#if __has_builtin(__builtin_amdgcn_global_load_async_to_lds_b128)
  typedef __attribute__((address_space(1))) v4i* gptr_t;
  typedef __attribute__((address_space(3))) v4i* lptr_t;
  __builtin_amdgcn_global_load_async_to_lds_b128(
      (gptr_t)(uintptr_t)gsrc,
      (lptr_t)(unsigned)(uintptr_t)lds_dst,
      0, 0);
#else
  const unsigned d = (unsigned)(uintptr_t)lds_dst;
  asm volatile("global_load_async_to_lds_b128 %0, %1, off"
               :: "v"(d), "v"(gsrc) : "memory");
#endif
}

template <int N> DEV void wait_async_n() {
#if __has_builtin(__builtin_amdgcn_s_wait_asynccnt)
  __builtin_amdgcn_s_wait_asynccnt(N);
#else
  asm volatile("s_wait_asynccnt %0" :: "i"(N) : "memory");
#endif
}

// ---------------------------------------------------------------------------
// K0: fp32 -> bf16 with scale
__global__ __launch_bounds__(256) void k_cvt(const float* __restrict__ in,
                                             bf16* __restrict__ out,
                                             float scale, size_t n) {
  size_t i = (size_t)blockIdx.x * blockDim.x + threadIdx.x;
  if (i < n) out[i] = (bf16)(in[i] * scale);
}

// ---------------------------------------------------------------------------
// K1: Y[M,N] = X[M,K] @ W[N,K]^T.  Block = 8 waves = 128 rows x 64 cols.
// W tile (64x32) staged via async-to-LDS, double-buffered; A fragment
// double-buffered in registers.
__global__ __launch_bounds__(256) void k_gemm_bf16(
    const bf16* __restrict__ A, const bf16* __restrict__ W,
    bf16* __restrict__ Cb, float* __restrict__ Cf, int M, int N, int K) {
  __shared__ __attribute__((aligned(16))) bf16 wtile[2][64 * 32];
  const int ntiles = N >> 6;
  const int nt   = (int)(blockIdx.x % ntiles);
  const int mblk = (int)(blockIdx.x / ntiles);
  const int wave = (int)(threadIdx.x >> 5);
  const size_t row0 = (size_t)mblk * 128 + (size_t)wave * 16;
  const size_t col0 = (size_t)nt * 64;

  // staging slice: 256 threads x 16B = 64 rows x 32 k of W
  const int tr = (int)(threadIdx.x >> 2);       // 0..63
  const int tc = (int)(threadIdx.x & 3) * 8;    // 0,8,16,24 (elements)

  {
    const bf16* src = W + (col0 + tr) * (size_t)K + tc;
    async_copy16(&wtile[0][tr * 32 + tc], src);
  }
  v16bf a = load_frag_a(A + row0 * K, K);
  v8f acc[4] = {};
  int p = 0;
  for (int k0 = 0; k0 < K; k0 += 32) {
    const bool more = (k0 + 32) < K;
    if (more) {
      const bf16* src = W + (col0 + tr) * (size_t)K + (k0 + 32) + tc;
      async_copy16(&wtile[p ^ 1][tr * 32 + tc], src);
    }
    v16bf an = a;
    if (more) an = load_frag_a(A + row0 * K + k0 + 32, K);
    if (more) wait_async_n<1>(); else wait_async_n<0>();
    __syncthreads();               // tile visible to all waves
#pragma unroll
    for (int j = 0; j < 4; ++j) {
      v16bf b = load_frag_b(&wtile[p][(j * 16) * 32], 32);
      acc[j] = wmma_bf16(a, b, acc[j]);
    }
    __syncthreads();               // all waves done reading before overwrite
    a = an;
    p ^= 1;
  }
#pragma unroll
  for (int j = 0; j < 4; ++j) {
    if (Cf) store_d_f32(Cf + row0 * N + col0 + j * 16, N, acc[j]);
    else    store_d_bf16(Cb + row0 * N + col0 + j * 16, N, acc[j]);
  }
}

// ---------------------------------------------------------------------------
// K2: S[b,h,k] = sum_q exp(logits[q,k]).  One wave per (b, h, 64-wide k
// tile); dp fragments are loop-invariant -> hoisted; loop over all 64
// q-tiles accumulating per-column exp sums in registers; plain store.
// (logits are O(1) here, so exp without max-subtraction is safe.)
__global__ __launch_bounds__(256) void k_colsum(
    const bf16* __restrict__ qp, const bf16* __restrict__ dp,
    float* __restrict__ S) {
  int gw = (int)((blockIdx.x * blockDim.x + threadIdx.x) >> 5);
  const int KT = LD_ / 64;
  const int kt = gw % KT; gw /= KT;
  const int h  = gw % H_; const int b = gw / H_;
  const bf16* dbase = dp + ((size_t)b * LD_ + (size_t)kt * 64) * D_ + h * DH_;

  v16bf bb[4][2];
#pragma unroll
  for (int j = 0; j < 4; ++j)
#pragma unroll
    for (int e = 0; e < 2; ++e)
      bb[j][e] = load_frag_b(dbase + (size_t)j * 16 * D_ + e * 32, D_);

  float sums[4] = {};
  for (int qt = 0; qt < LQ_ / 16; ++qt) {
    const bf16* qbase =
        qp + ((size_t)b * LQ_ + (size_t)qt * 16) * D_ + h * DH_;
    v16bf a0 = load_frag_a(qbase, D_);
    v16bf a1 = load_frag_a(qbase + 32, D_);
    v8f acc[4] = {};
#pragma unroll
    for (int j = 0; j < 4; ++j) {
      acc[j] = wmma_bf16(a0, bb[j][0], acc[j]);
      acc[j] = wmma_bf16(a1, bb[j][1], acc[j]);
    }
#pragma unroll
    for (int j = 0; j < 4; ++j)
#pragma unroll
      for (int r = 0; r < 8; ++r) sums[j] += __expf(acc[j][r]);
  }
  const int l = lane_id(), n = l & 15;
  float* srow = S + ((size_t)b * H_ + h) * LD_ + (size_t)kt * 64;
#pragma unroll
  for (int j = 0; j < 4; ++j) {
    float s = sums[j];
    s += __shfl_xor(s, 16, 32);   // fold the two 8-row halves per column
    if (l < 16) srow[j * 16 + n] = s;
  }
}

// ---------------------------------------------------------------------------
// K4: vsT[b][e][k] = dp[b][k][e] / S[b][e/64][k]
__global__ __launch_bounds__(256) void k_scale_v(
    const bf16* __restrict__ dp, const float* __restrict__ S,
    bf16* __restrict__ vsT) {
  size_t i = (size_t)blockIdx.x * blockDim.x + threadIdx.x;
  const size_t total = (size_t)B_ * D_ * LD_;
  if (i >= total) return;
  const int k = (int)(i % LD_);
  const int e = (int)((i / LD_) % D_);
  const int b = (int)(i / ((size_t)LD_ * D_));
  const float v = (float)dp[((size_t)b * LD_ + k) * D_ + e];
  const float s = S[((size_t)b * H_ + (e / DH_)) * LD_ + k];
  vsT[i] = (bf16)(v / s);
}

// ---------------------------------------------------------------------------
// K5: ctx[b,q,h*64+e] = sum_k exp(l[q,k]) * vsT[b][h*64+e][k].
// Block = 8 waves = one (b,h), 128 q-rows.  Per 128-key step the block
// async-stages dp (128x64) and vsT (64x128) tiles into double-buffered LDS;
// each wave then runs 16 logit WMMAs, exp -> per-wave P staging, and 16
// context WMMAs entirely out of LDS.
// Dynamic LDS: 2*16KB (dp) + 2*16KB (vsT) + 8*4KB (P) = 96 KB.
__global__ __launch_bounds__(256) void k_ctx(
    const bf16* __restrict__ qp, const bf16* __restrict__ dp,
    const bf16* __restrict__ vsT, bf16* __restrict__ ctx) {
  extern __shared__ __attribute__((aligned(16))) char smem[];
  bf16* dt = (bf16*)smem;                             // [2][128*64]
  bf16* vt = (bf16*)(smem + 2 * 128 * 64 * 2);        // [2][64*128]
  bf16* pb = (bf16*)(smem + 4 * 128 * 64 * 2);        // [8][16*128]

  int gb = (int)blockIdx.x;
  const int QB = LQ_ / 128;
  const int qblk = gb % QB; gb /= QB;
  const int h = gb % H_; const int b = gb / H_;
  const int wave = (int)(threadIdx.x >> 5);
  const int t = (int)threadIdx.x;
  const int qt = qblk * 8 + wave;

  const bf16* dpb = dp + (size_t)b * LD_ * D_ + h * DH_;      // + k*D
  const bf16* vsb = vsT + ((size_t)b * D_ + h * DH_) * LD_;   // + e*LD + k
  bf16* buf = pb + (size_t)wave * 16 * 128;

  const bf16* qbase = qp + ((size_t)b * LQ_ + (size_t)qt * 16) * D_ + h * DH_;
  const int l = lane_id(), n = l & 15, mh = (l >> 4) * 8;

  v16bf aq[2];
  aq[0] = load_frag_a(qbase, D_);
  aq[1] = load_frag_a(qbase + 32, D_);

  // 8 async 16B copies per thread per stage: 16KB dp tile + 16KB vsT tile.
  auto stage = [&](int pb_, int k0) {
    bf16* dtd = dt + (size_t)pb_ * (128 * 64);
    bf16* vtd = vt + (size_t)pb_ * (64 * 128);
    const bf16* ds = dpb + (size_t)k0 * D_;
    const bf16* vs = vsb + k0;
#pragma unroll
    for (int i = 0; i < 4; ++i) {           // dp: [krow 0..127][e 0..63]
      const int c = t + 256 * i;
      const int r = c >> 3, col = (c & 7) * 8;
      async_copy16(dtd + (size_t)r * 64 + col, ds + (size_t)r * D_ + col);
    }
#pragma unroll
    for (int i = 0; i < 4; ++i) {           // vsT: [e 0..63][k 0..127]
      const int c = t + 256 * i;
      const int r = c >> 4, col = (c & 15) * 8;
      async_copy16(vtd + (size_t)r * 128 + col, vs + (size_t)r * LD_ + col);
    }
  };

  stage(0, 0);
  v8f acc[4] = {};
  int p = 0;
  for (int k0 = 0; k0 < LD_; k0 += 128) {
    const bool more = (k0 + 128) < LD_;
    if (more) stage(p ^ 1, k0 + 128);
    if (more) wait_async_n<8>(); else wait_async_n<0>();
    __syncthreads();                         // tiles visible to all waves
    const bf16* dtp = dt + (size_t)p * (128 * 64);
    const bf16* vtp = vt + (size_t)p * (64 * 128);
    // ---- P[16 q x 128 k] = exp(logits), chunked 16x32, all from LDS -----
#pragma unroll
    for (int kk = 0; kk < 128; kk += 32) {
      v8f lg[2] = {};
#pragma unroll
      for (int half = 0; half < 2; ++half)
#pragma unroll
        for (int e0 = 0; e0 < 2; ++e0) {
          v16bf bb = load_frag_b(
              dtp + (size_t)(kk + half * 16) * 64 + e0 * 32, 64);
          lg[half] = wmma_bf16(aq[e0], bb, lg[half]);
        }
#pragma unroll
      for (int half = 0; half < 2; ++half)
#pragma unroll
        for (int r = 0; r < 8; ++r)
          buf[(size_t)(mh + r) * 128 + kk + half * 16 + n] =
              (bf16)__expf(lg[half][r]);
    }
    asm volatile("s_wait_dscnt 0" ::: "memory");   // own P stores complete
    // ---- ctx += P @ vsT tile, all from LDS ------------------------------
#pragma unroll
    for (int kk = 0; kk < 128; kk += 32) {
      v16bf pa = load_frag_a(buf + kk, 128);
#pragma unroll
      for (int j = 0; j < 4; ++j) {
        v16bf vbf = load_frag_b(vtp + (size_t)(j * 16) * 128 + kk, 128);
        acc[j] = wmma_bf16(pa, vbf, acc[j]);
      }
    }
    __syncthreads();                         // done reading before overwrite
    p ^= 1;
  }
#pragma unroll
  for (int j = 0; j < 4; ++j)
    store_d_bf16(ctx + ((size_t)b * LQ_ + (size_t)qt * 16) * D_ + h * DH_ +
                     (size_t)j * 16,
                 D_, acc[j]);
}

// ---------------------------------------------------------------------------
extern "C" void kernel_launch(void* const* d_in, const int* in_sizes, int n_in,
                              void* d_out, int out_size, void* d_ws,
                              size_t ws_size, hipStream_t stream) {
  (void)in_sizes; (void)n_in; (void)out_size; (void)ws_size;
  const float* queries = (const float*)d_in[0];
  const float* doc     = (const float*)d_in[1];
  const float* Wt      = (const float*)d_in[2];
  const float* Wo      = (const float*)d_in[3];
  float* out = (float*)d_out;

  char* ws = (char*)d_ws;
  size_t off = 0;
  auto alloc = [&](size_t bytes) {
    char* p = ws + off;
    off += (bytes + 255) & ~(size_t)255;
    return p;
  };
  bf16*  qbf   = (bf16*)alloc((size_t)B_ * LQ_ * D_ * 2);
  bf16*  docbf = (bf16*)alloc((size_t)B_ * LD_ * D_ * 2);
  bf16*  Wtbf  = (bf16*)alloc((size_t)D_ * D_ * 2);
  bf16*  Wobf  = (bf16*)alloc((size_t)D_ * D_ * 2);
  bf16*  qp    = (bf16*)alloc((size_t)B_ * LQ_ * D_ * 2);
  bf16*  dp    = (bf16*)alloc((size_t)B_ * LD_ * D_ * 2);
  float* S     = (float*)alloc((size_t)B_ * H_ * LD_ * 4);
  bf16* vsT = docbf;  // reuse: dead after projections
  bf16* ctx = qbf;    // reuse: dead after projections

  auto cvt = [&](const float* in, bf16* o, float sc, size_t n) {
    k_cvt<<<(unsigned)((n + 255) / 256), 256, 0, stream>>>(in, o, sc, n);
  };
  cvt(queries, qbf, QUERY_SCALE_, (size_t)B_ * LQ_ * D_);
  cvt(doc, docbf, 1.f, (size_t)B_ * LD_ * D_);
  cvt(Wt, Wtbf, 1.f, (size_t)D_ * D_);
  cvt(Wo, Wobf, 1.f, (size_t)D_ * D_);

  auto gemm = [&](const bf16* A, const bf16* W, bf16* Cb, float* Cf, int M,
                  int N, int K) {
    const unsigned blocks = (unsigned)((M / 128) * (N / 64));
    k_gemm_bf16<<<blocks, 256, 0, stream>>>(A, W, Cb, Cf, M, N, K);
  };
  gemm(qbf,   Wtbf, qp, nullptr, B_ * LQ_, D_, D_);
  gemm(docbf, Wtbf, dp, nullptr, B_ * LD_, D_, D_);

  {
    const int waves = B_ * H_ * (LD_ / 64);
    k_colsum<<<waves / 8, 256, 0, stream>>>(qp, dp, S);
  }
  {
    const size_t n = (size_t)B_ * D_ * LD_;
    k_scale_v<<<(unsigned)((n + 255) / 256), 256, 0, stream>>>(dp, S, vsT);
  }
  {
    const unsigned blocks = (unsigned)(B_ * H_ * (LQ_ / 128));
    k_ctx<<<blocks, 256, 96 * 1024, stream>>>(qp, dp, vsT, ctx);
  }
  gemm(ctx, Wobf, nullptr, out, B_ * LQ_, D_, D_);
}
